// HoughOnTokens_26053271617623
// MI455X (gfx1250) — matrix operand order repaired and checked by reading, not compiled, not run on hardware
//
#include <hip/hip_runtime.h>
#include <math.h>

// Problem constants (from reference: B=32, 32x32 tokens, D=768, C=64, ch=32,
// theta_res=3 -> n_theta=60, rho_res=1 -> n_rho=89, HR=5340).
#define HR     5340
#define HRP    5344      // HR padded to multiple of 32
#define NTOK   1024
#define BB     32
#define DMOD   768
#define CC     64
#define CH     32
#define MROWS  32768     // B * NTOK
#define PADR   240       // 4 * n_theta rho-conv halo
#define HTROWS 5824      // PADR + HRP + PADR (covers junk-tile reads)

typedef __attribute__((ext_vector_type(16))) __bf16 v16bf;
typedef __attribute__((ext_vector_type(8)))  __bf16 v8bf;
typedef __attribute__((ext_vector_type(8)))  float  v8f;

static __device__ __forceinline__ __bf16 to_bf(float x) { return (__bf16)x; }

static __device__ __forceinline__ v8f wmma_bf(v16bf a, v16bf b, v8f c) {
  return __builtin_amdgcn_wmma_f32_16x16x32_bf16(false, a, false, b, (short)0, c,
                                                 false, false);
}

static __device__ __forceinline__ v8f splat8(float v) {
  v8f r;
#pragma unroll
  for (int i = 0; i < 8; ++i) r[i] = v;
  return r;
}

static __device__ __forceinline__ v16bf cat16(v8bf lo, v8bf hi) {
  return __builtin_shufflevector(lo, hi, 0, 1, 2, 3, 4, 5, 6, 7,
                                 8, 9, 10, 11, 12, 13, 14, 15);
}

// ---- bf16 fragment loaders (no conversions: raw 16B vector loads) ---------
// A fragment (16 rows x 32 K) from row-major bf16 [*, lda] (lda mult of 8).
// Lane L holds row m=L&15; lanes<16: K 0-7 & 16-23, lanes>=16: K 8-15 & 24-31.
static __device__ __forceinline__ v16bf load_a_bf(const __bf16* __restrict__ src,
                                                  int row0, int k0, int lda,
                                                  int lane) {
  const int m  = lane & 15;
  const int kb = (lane < 16) ? 0 : 8;
  const __bf16* p = src + (size_t)(row0 + m) * (size_t)lda + k0 + kb;
  const v8bf lo = *(const v8bf*)(p);
  const v8bf hi = *(const v8bf*)(p + 16);
  return cat16(lo, hi);
}

// B fragment (32 K x 16 cols) from "weight-style" row-major bf16 W[n, k].
// Lane L holds col n=L&15; lanes<16: K 0-7 & 16-23, lanes>=16: K 8-15 & 24-31.
static __device__ __forceinline__ v16bf load_b_bf(const __bf16* __restrict__ W,
                                                  int col0, int k0, int lda,
                                                  int lane) {
  const int n  = col0 + (lane & 15);
  const int kb = (lane < 16) ? 0 : 8;
  const __bf16* p = W + (size_t)n * (size_t)lda + k0 + kb;
  const v8bf lo = *(const v8bf*)(p);
  const v8bf hi = *(const v8bf*)(p + 16);
  return cat16(lo, hi);
}

// Strided bf16 B fragment (for W2[co, ci, 9]: n-stride 288, k-stride 9).
static __device__ __forceinline__ v16bf load_b_bf_str(const __bf16* __restrict__ W,
                                                      int col0, int ldn, int ldk,
                                                      int lane) {
  const int n  = col0 + (lane & 15);
  const int kb = (lane < 16) ? 0 : 8;
  const __bf16* p = W + (size_t)n * ldn + (size_t)kb * ldk;
  v16bf b;
#pragma unroll
  for (int e = 0; e < 16; ++e) b[e] = p[(e + ((e >= 8) ? 8 : 0)) * ldk];
  return b;
}

// ---- f32 fragment loaders (single-use operands: convert on the fly) -------
static __device__ __forceinline__ v16bf load_a_rm(const float* __restrict__ src,
                                                  int row0, int k0, int lda,
                                                  int lane) {
  const int m  = lane & 15;
  const int kb = (lane < 16) ? 0 : 8;
  const float* p = src + (size_t)(row0 + m) * (size_t)lda + k0 + kb;
  v16bf a;
#pragma unroll
  for (int g = 0; g < 2; ++g) {
    const float4 x = *(const float4*)(p + g * 16);
    const float4 y = *(const float4*)(p + g * 16 + 4);
    const int e = g * 8;
    a[e + 0] = to_bf(x.x); a[e + 1] = to_bf(x.y);
    a[e + 2] = to_bf(x.z); a[e + 3] = to_bf(x.w);
    a[e + 4] = to_bf(y.x); a[e + 5] = to_bf(y.y);
    a[e + 6] = to_bf(y.z); a[e + 7] = to_bf(y.w);
  }
  return a;
}

// ---------------------------------------------------------------------------
__global__ void k_zero(float* __restrict__ p, size_t n) {
  size_t i = (size_t)blockIdx.x * blockDim.x + threadIdx.x;
  const size_t stride = (size_t)gridDim.x * blockDim.x;
  for (; i < n; i += stride) p[i] = 0.f;
}

__global__ void k_pack(const float* __restrict__ s, __bf16* __restrict__ d, int n) {
  const int i = blockIdx.x * 256 + threadIdx.x;
  if (i < n) d[i] = (__bf16)s[i];
}

__global__ void k_norm(const float* __restrict__ vote, float* __restrict__ norminv) {
  const int h = blockIdx.x * blockDim.x + threadIdx.x;
  if (h >= HRP) return;
  if (h >= HR) { norminv[h] = 0.f; return; }
  float s = 0.f;
  for (int n = 0; n < NTOK; ++n) s += vote[(size_t)n * HR + h];
  norminv[h] = 1.f / fmaxf(s, 1.f);
}

// vote -> bf16, both transposed [h][n] (HT B operand) and padded row-major
// [n][HRP] (IHT A operand; pad columns zeroed so the K loop needs no guards).
__global__ void k_votet(const float* __restrict__ vote,
                        __bf16* __restrict__ voteT,
                        __bf16* __restrict__ voteB) {
  const int h = blockIdx.x * 64 + threadIdx.x;
  const int n = blockIdx.y * 4 + threadIdx.y;
  if (h >= HRP) return;
  const float v = (h < HR) ? vote[(size_t)n * HR + h] : 0.f;
  const __bf16 bv = (__bf16)v;
  voteT[(size_t)h * NTOK + n] = bv;
  voteB[(size_t)n * HRP + h] = bv;
}

// to_feats (+bias) then conv1 + relu; stores feat f32 [M,64] (residual) and
// y transposed as bf16 yT[b*32+c][n] for the HT GEMM.
__global__ __launch_bounds__(128) void k_feats(const float* __restrict__ tokens,
                                               const __bf16* __restrict__ wtinB,
                                               const float* __restrict__ bt_in,
                                               const __bf16* __restrict__ w1B,
                                               float* __restrict__ feat,
                                               __bf16* __restrict__ yT) {
  __shared__ float lds[4][16][68];
  const int lane = threadIdx.x & 31;
  const int wave = threadIdx.x >> 5;
  const int row0 = (blockIdx.x * 4 + wave) * 16;
  const int cn   = lane & 15;

  v8f acc[4];
#pragma unroll
  for (int nt = 0; nt < 4; ++nt) acc[nt] = splat8(bt_in[nt * 16 + cn]);

  for (int k0 = 0; k0 < DMOD; k0 += 32) {
    v16bf a = load_a_rm(tokens, row0, k0, DMOD, lane);
#pragma unroll
    for (int nt = 0; nt < 4; ++nt) {
      v16bf b = load_b_bf(wtinB, nt * 16, k0, DMOD, lane);
      acc[nt] = wmma_bf(a, b, acc[nt]);
    }
  }

  float* ws = &lds[wave][0][0];
#pragma unroll
  for (int nt = 0; nt < 4; ++nt) {
    const int col = nt * 16 + cn;
#pragma unroll
    for (int j = 0; j < 8; ++j) {
      const int m = (lane < 16) ? j : (8 + j);
      const float v = acc[nt][j];
      feat[(size_t)(row0 + m) * CC + col] = v;
      ws[m * 68 + col] = v;
    }
  }
  __syncthreads();

  v8f acc2[2] = { splat8(0.f), splat8(0.f) };
#pragma unroll
  for (int ks = 0; ks < 2; ++ks) {
    v16bf a = load_a_rm(ws, 0, ks * 32, 68, lane);
#pragma unroll
    for (int nt = 0; nt < 2; ++nt) {
      v16bf b = load_b_bf(w1B, nt * 16, ks * 32, CC, lane);
      acc2[nt] = wmma_bf(a, b, acc2[nt]);
    }
  }
  const int bimg = row0 >> 10;
  const int tok0 = row0 & 1023;
#pragma unroll
  for (int nt = 0; nt < 2; ++nt) {
    const int c = nt * 16 + cn;
    __bf16* dst = yT + ((size_t)(bimg * CH + c) << 10) + tok0 + ((lane < 16) ? 0 : 8);
#pragma unroll
    for (int j = 0; j < 8; ++j) dst[j] = to_bf(fmaxf(acc2[nt][j], 0.f));
  }
}

// HT: D[(b,c), h] = sum_n yT[(b,c), n] * voteT[h, n], scaled by 1/norm[h].
// Output bf16, stored [b][h + PADR][c] (rho-padded) for the (9,1) conv.
__global__ __launch_bounds__(128) void k_ht(const __bf16* __restrict__ yT,
                                            const __bf16* __restrict__ voteT,
                                            const float* __restrict__ norminv,
                                            __bf16* __restrict__ htTp) {
  const int lane = threadIdx.x & 31;
  const int wave = threadIdx.x >> 5;
  const int row0 = (blockIdx.y * 4 + wave) * 16;   // over B*CH = 1024
  const int h0   = blockIdx.x * 64;
  const int cn   = lane & 15;

  v8f acc[4] = { splat8(0.f), splat8(0.f), splat8(0.f), splat8(0.f) };
  for (int k0 = 0; k0 < NTOK; k0 += 32) {
    v16bf a = load_a_bf(yT, row0, k0, NTOK, lane);
#pragma unroll
    for (int nt = 0; nt < 4; ++nt) {
      const int hb = h0 + nt * 16;
      if (hb < HRP) {                 // wave-uniform guard: EXEC stays full
        v16bf b = load_b_bf(voteT, hb, k0, NTOK, lane);
        acc[nt] = wmma_bf(a, b, acc[nt]);
      }
    }
  }
  const int bimg = row0 >> 5;
  const int c0   = (row0 & 31) + ((lane < 16) ? 0 : 8);
#pragma unroll
  for (int nt = 0; nt < 4; ++nt) {
    const int h = h0 + nt * 16 + cn;
    if (h < HR) {
      const float nv = norminv[h];
      __bf16* dst = htTp + ((size_t)bimg * HTROWS + (h + PADR)) * CH + c0;
#pragma unroll
      for (int j = 0; j < 8; ++j) dst[j] = to_bf(acc[nt][j] * nv);
    }
  }
}

// (9,1) conv along rho as 9 accumulated WMMAs over ci; relu on store.
// Output bf16 stored [b][co][h] (K-contiguous for the IHT B operand).
__global__ __launch_bounds__(128) void k_conv(const __bf16* __restrict__ htTp,
                                              const __bf16* __restrict__ w2B,
                                              __bf16* __restrict__ convT) {
  const int lane = threadIdx.x & 31;
  const int wave = threadIdx.x >> 5;
  const int h0 = (blockIdx.x * 4 + wave) * 16;
  if (h0 >= HRP) return;
  const int b = blockIdx.y;
  const __bf16* hb = htTp + (size_t)b * HTROWS * CH;

  v8f acc[2] = { splat8(0.f), splat8(0.f) };
#pragma unroll
  for (int dr = 0; dr < 9; ++dr) {
    const int p0 = h0 + PADR + (dr - 4) * 60;      // shift by whole rho rows
    v16bf a = load_a_bf(hb, p0, 0, CH, lane);
#pragma unroll
    for (int nt = 0; nt < 2; ++nt) {
      v16bf w = load_b_bf_str(w2B + dr, nt * 16, CH * 9, 9, lane);
      acc[nt] = wmma_bf(a, w, acc[nt]);
    }
  }
  const int hme = (lane < 16) ? 0 : 8;
#pragma unroll
  for (int nt = 0; nt < 2; ++nt) {
    const int co = nt * 16 + (lane & 15);
    __bf16* dst = convT + (size_t)(b * CH + co) * HRP;
#pragma unroll
    for (int j = 0; j < 8; ++j) {
      const int h = h0 + hme + j;
      if (h < HR) dst[h] = to_bf(fmaxf(acc[nt][j], 0.f));
    }
  }
}

// IHT: iht[b,n,c] = (1/60) * sum_h vote[n,h] * relu(conv)[b,c,h]; bf16 out.
__global__ __launch_bounds__(128) void k_iht(const __bf16* __restrict__ voteB,
                                             const __bf16* __restrict__ convT,
                                             __bf16* __restrict__ iht) {
  const int lane = threadIdx.x & 31;
  const int wave = threadIdx.x >> 5;
  const int row0 = (blockIdx.x * 4 + wave) * 16;   // token rows
  const int b = blockIdx.y;
  const __bf16* cb = convT + (size_t)b * CH * HRP;

  v8f acc[2] = { splat8(0.f), splat8(0.f) };
  for (int k0 = 0; k0 < HRP; k0 += 32) {
    v16bf a = load_a_bf(voteB, row0, k0, HRP, lane);
#pragma unroll
    for (int nt = 0; nt < 2; ++nt) {
      v16bf bb = load_b_bf(cb, nt * 16, k0, HRP, lane);
      acc[nt] = wmma_bf(a, bb, acc[nt]);
    }
  }
  const float sc = 1.f / 60.f;
#pragma unroll
  for (int nt = 0; nt < 2; ++nt) {
    const int c = nt * 16 + (lane & 15);
    __bf16* dst = iht + ((size_t)b * NTOK + row0) * CH + c;
#pragma unroll
    for (int j = 0; j < 8; ++j) {
      const int m = (lane < 16) ? j : (8 + j);
      dst[(size_t)m * CH] = to_bf(acc[nt][j] * sc);
    }
  }
}

// conv3, residual add with feat, then from_feats (+bias) -> out [M, 768].
__global__ __launch_bounds__(128) void k_out(const __bf16* __restrict__ ihtB,
                                             const __bf16* __restrict__ w3B,
                                             const float* __restrict__ feat,
                                             const __bf16* __restrict__ wtoutB,
                                             const float* __restrict__ bt_out,
                                             float* __restrict__ out) {
  __shared__ float lds[4][16][68];
  const int lane = threadIdx.x & 31;
  const int wave = threadIdx.x >> 5;
  const int row0 = (blockIdx.x * 4 + wave) * 16;
  const int cn   = lane & 15;

  v8f acc[4];
#pragma unroll
  for (int nt = 0; nt < 4; ++nt) acc[nt] = splat8(0.f);
  {
    v16bf a = load_a_bf(ihtB, row0, 0, CH, lane);  // K = 32, single step
#pragma unroll
    for (int nt = 0; nt < 4; ++nt) {
      v16bf b = load_b_bf(w3B, nt * 16, 0, CH, lane);
      acc[nt] = wmma_bf(a, b, acc[nt]);
    }
  }
  float* ws = &lds[wave][0][0];
#pragma unroll
  for (int nt = 0; nt < 4; ++nt) {
    const int col = nt * 16 + cn;
#pragma unroll
    for (int j = 0; j < 8; ++j) {
      const int m = (lane < 16) ? j : (8 + j);
      ws[m * 68 + col] = acc[nt][j] + feat[(size_t)(row0 + m) * CC + col];
    }
  }
  __syncthreads();

  v16bf a2[2];
#pragma unroll
  for (int ks = 0; ks < 2; ++ks) a2[ks] = load_a_rm(ws, 0, ks * 32, 68, lane);

  for (int ot = 0; ot < 48; ++ot) {
    const int col0 = ot * 16;
    v8f acc2 = splat8(bt_out[col0 + cn]);
#pragma unroll
    for (int ks = 0; ks < 2; ++ks) {
      v16bf b = load_b_bf(wtoutB, col0, ks * 32, CC, lane);
      acc2 = wmma_bf(a2[ks], b, acc2);
    }
#pragma unroll
    for (int j = 0; j < 8; ++j) {
      const int m = (lane < 16) ? j : (8 + j);
      out[(size_t)(row0 + m) * DMOD + col0 + cn] = acc2[j];
    }
  }
}

// ---------------------------------------------------------------------------
extern "C" void kernel_launch(void* const* d_in, const int* in_sizes, int n_in,
                              void* d_out, int out_size, void* d_ws, size_t ws_size,
                              hipStream_t stream) {
  (void)in_sizes; (void)n_in; (void)out_size; (void)ws_size;
  const float* tokens = (const float*)d_in[0];
  const float* vote   = (const float*)d_in[1];
  const float* Wt_in  = (const float*)d_in[2];
  const float* bt_in  = (const float*)d_in[3];
  const float* W1     = (const float*)d_in[4];
  const float* W2     = (const float*)d_in[5];
  const float* W3     = (const float*)d_in[6];
  const float* Wt_out = (const float*)d_in[7];
  const float* bt_out = (const float*)d_in[8];
  float* out = (float*)d_out;
  char* ws = (char*)d_ws;

  // Workspace layout (byte offsets, all 128B-aligned), ~57.6 MB total.
  const size_t o_norm  = 0;                                   // f32 [HRP]
  const size_t o_voteT = 21504;                               // bf16 [HRP][NTOK]
  const size_t o_voteB = o_voteT + (size_t)HRP * NTOK * 2;    // bf16 [NTOK][HRP]
  const size_t o_feat  = o_voteB + (size_t)NTOK * HRP * 2;    // f32  [MROWS][CC]
  const size_t o_yT    = o_feat + (size_t)MROWS * CC * 4;     // bf16 [NTOK][NTOK]
  const size_t o_wtin  = o_yT + (size_t)NTOK * NTOK * 2;      // bf16 [CC][DMOD]
  const size_t o_w1    = o_wtin + (size_t)CC * DMOD * 2;      // bf16 [CH][CC]
  const size_t o_w2    = o_w1 + (size_t)CH * CC * 2;          // bf16 [CH*CH*9]
  const size_t o_w3    = o_w2 + (size_t)CH * CH * 9 * 2 + 128;// bf16 [CC][CH]
  const size_t o_wtout = o_w3 + (size_t)CC * CH * 2;          // bf16 [DMOD][CC]
  const size_t o_htTp  = o_wtout + (size_t)DMOD * CC * 2;     // bf16 [BB][HTROWS][CH]
  const size_t o_convT = o_htTp + (size_t)BB * HTROWS * CH * 2;// bf16 [BB][CH][HRP]
  const size_t o_iht   = o_convT + (size_t)BB * CH * HRP * 2; // bf16 [BB][NTOK][CH]

  float*  norminv = (float*)(ws + o_norm);
  __bf16* voteT   = (__bf16*)(ws + o_voteT);
  __bf16* voteB   = (__bf16*)(ws + o_voteB);
  float*  feat    = (float*)(ws + o_feat);
  __bf16* yT      = (__bf16*)(ws + o_yT);
  __bf16* wtinB   = (__bf16*)(ws + o_wtin);
  __bf16* w1B     = (__bf16*)(ws + o_w1);
  __bf16* w2B     = (__bf16*)(ws + o_w2);
  __bf16* w3B     = (__bf16*)(ws + o_w3);
  __bf16* wtoutB  = (__bf16*)(ws + o_wtout);
  __bf16* htTp    = (__bf16*)(ws + o_htTp);
  __bf16* convT   = (__bf16*)(ws + o_convT);
  __bf16* ihtB    = (__bf16*)(ws + o_iht);

  // Zero the rho-padded hough buffer + conv output (bf16, dword fill).
  const size_t zero_dw = ((size_t)BB * HTROWS * CH * 2 + (size_t)BB * CH * HRP * 2) / 4;
  k_zero<<<2048, 256, 0, stream>>>((float*)htTp, zero_dw);
  k_norm<<<(HRP + 255) / 256, 256, 0, stream>>>(vote, norminv);
  k_votet<<<dim3((HRP + 63) / 64, NTOK / 4), dim3(64, 4), 0, stream>>>(vote, voteT, voteB);
  k_pack<<<(CC * DMOD + 255) / 256, 256, 0, stream>>>(Wt_in, wtinB, CC * DMOD);
  k_pack<<<(CH * CC + 255) / 256, 256, 0, stream>>>(W1, w1B, CH * CC);
  k_pack<<<(CH * CH * 9 + 255) / 256, 256, 0, stream>>>(W2, w2B, CH * CH * 9);
  k_pack<<<(CC * CH + 255) / 256, 256, 0, stream>>>(W3, w3B, CC * CH);
  k_pack<<<(DMOD * CC + 255) / 256, 256, 0, stream>>>(Wt_out, wtoutB, DMOD * CC);
  k_feats<<<MROWS / 64, 128, 0, stream>>>(tokens, wtinB, bt_in, w1B, feat, yT);
  k_ht<<<dim3(84, 16), 128, 0, stream>>>(yT, voteT, norminv, htTp);
  k_conv<<<dim3(84, BB), 128, 0, stream>>>(htTp, w2B, convT);
  k_iht<<<dim3(16, BB), 128, 0, stream>>>(voteB, convT, ihtB);
  k_out<<<MROWS / 64, 128, 0, stream>>>(ihtB, w3B, feat, wtoutB, bt_out, out);
}
